// SceneDecodeLayer_37452114821366
// MI455X (gfx1250) — matrix operand order repaired and checked by reading, not compiled, not run on hardware
//
#include <hip/hip_runtime.h>
#include <hip/hip_bf16.h>

typedef __bf16 v16bf __attribute__((ext_vector_type(16)));
typedef __bf16 v8bf  __attribute__((ext_vector_type(8)));
typedef float  v8f   __attribute__((ext_vector_type(8)));

#define BATCH 2
#define CCH   64
#define HH    256
#define WWI   256
#define NPIX  (BATCH * HH * WWI)   // 131072

// ---------------------------------------------------------------------------
// Wave32 all-lane sum reduction
// ---------------------------------------------------------------------------
__device__ __forceinline__ float wave_sum32(float v) {
  #pragma unroll
  for (int off = 16; off > 0; off >>= 1) v += __shfl_xor(v, off, 32);
  return v;
}

// ---------------------------------------------------------------------------
// gfx1250 async copy global->LDS (16 bytes per enabled lane), ASYNCcnt-tracked.
// lds_off is the wave-relative LDS byte address (low 32 bits of a generic LDS
// pointer, per the ISA flat->LDS truncation rule).
// ---------------------------------------------------------------------------
__device__ __forceinline__ void async_g2l_b128(unsigned lds_off, const void* g) {
  asm volatile("global_load_async_to_lds_b128 %0, %1, off"
               :: "v"(lds_off), "v"(g) : "memory");
}
__device__ __forceinline__ void wait_asynccnt0() {
  asm volatile("s_wait_asynccnt 0x0" ::: "memory");
}

// ---------------------------------------------------------------------------
// Kernel 1: repack conv weights (OIHW f32) into WMMA A-fragment order (bf16).
// A is 16x32 (MxK) per k-step. ISA layout (16-bit A 16x32):
//   lane<16 : M=lane,    halves 0-7 -> K 0..7,  halves 8-15 -> K 16..23
//   lane>=16: M=lane-16, halves 0-7 -> K 8..15, halves 8-15 -> K 24..31
// K ordering: k = s*CIN + ci  (s = kh*3+kw).
// frag index = ((mt*KSTEPS + ks)*32 + lane)*16 + t
// ---------------------------------------------------------------------------
__device__ __forceinline__ void pack_one(const float* __restrict__ W,
                                         __bf16* __restrict__ frag,
                                         int idx, int CIN, int KSTEPS) {
  int t    = idx & 15;
  int lane = (idx >> 4) & 31;
  int rest = idx >> 9;                 // mt*KSTEPS + ks
  int ks   = rest % KSTEPS;
  int mt   = rest / KSTEPS;
  int M    = (mt << 4) + (lane & 15);
  int Koff = (t & 7) + ((lane >> 4) << 3) + ((t >> 3) << 4);
  int K    = (ks << 5) + Koff;
  int s    = K / CIN;
  int ci   = K % CIN;
  frag[idx] = (__bf16)W[(M * CIN + ci) * 9 + s];
}

__global__ __launch_bounds__(256) void repack_weights_kernel(
    const float* __restrict__ Wa, const float* __restrict__ Wh1,
    __bf16* __restrict__ fragA, __bf16* __restrict__ fragH) {
  int idx = blockIdx.x * 256 + threadIdx.x;
  const int NA = 4 * 36 * 512;         // conv1: Cin=128 -> 36 k-steps
  if (idx < NA) pack_one(Wa, fragA, idx, 128, 36);
  else          pack_one(Wh1, fragH, idx - NA, 64, 18);
}

// ---------------------------------------------------------------------------
// Kernel 2: fused trilinear sampling + per-pixel softmax attention.
// One wave per pixel, 2 channels per lane (c = lane, lane+32).
// Emits xcat = concat(attn_out, feat_j) as NHWC bf16 (128 ch / pixel).
// ---------------------------------------------------------------------------
__global__ __launch_bounds__(256) void attn_sample_kernel(
    const float* __restrict__ feat_j, const float* __restrict__ feat_p,
    const float* __restrict__ kv_j,   const float* __restrict__ kv_p,
    const float* __restrict__ grids,  const float* __restrict__ beta,
    __bf16* __restrict__ xcat) {
  int tid  = blockIdx.x * 256 + threadIdx.x;
  int lane = tid & 31;
  int pix  = tid >> 5;
  int w = pix & 255, h = (pix >> 8) & 255, b = pix >> 16;
  int hw = (h << 8) | w;
  int c0 = lane, c1 = lane + 32;

  float fj0 = feat_j[((b * CCH + c0) << 16) + hw];
  float fj1 = feat_j[((b * CCH + c1) << 16) + hw];
  float fp0 = feat_p[((b * CCH + c0) << 16) + hw];
  float fp1 = feat_p[((b * CCH + c1) << 16) + hw];
  float bet = beta[0];

  float sj[3][2];
  float logit[3];

  #pragma unroll
  for (int r = 0; r < 3; ++r) {
    int D = r + 1;
    long gbase = (long)(((b * 3 + r) * 256 + h) * 256 + w) * 3;
    float gx = grids[gbase + 0] * 2.f - 1.f;
    float gy = grids[gbase + 1] * 2.f - 1.f;
    float gz = grids[gbase + 2] * 2.f - 1.f;
    float x = (gx + 1.f) * 0.5f * (WWI - 1);
    float y = (gy + 1.f) * 0.5f * (HH - 1);
    float z = (gz + 1.f) * 0.5f * (float)(D - 1);
    float x0f = floorf(x), y0f = floorf(y), z0f = floorf(z);
    float wx = x - x0f, wy = y - y0f, wz = z - z0f;
    int x0 = min(max((int)x0f, 0), WWI - 1);
    int x1 = min(max((int)x0f + 1, 0), WWI - 1);
    int y0 = min(max((int)y0f, 0), HH - 1);
    int y1 = min(max((int)y0f + 1, 0), HH - 1);
    int z0 = min(max((int)z0f, 0), D - 1);
    int z1 = min(max((int)z0f + 1, 0), D - 1);

    int i00 = (y0 << 8) | x0, i01 = (y0 << 8) | x1;
    int i10 = (y1 << 8) | x0, i11 = (y1 << 8) | x1;

    auto tri = [&](const float* __restrict__ vol, int c) -> float {
      int b0 = ((b * 3 + z0) * CCH + c) << 16;
      int b1 = ((b * 3 + z1) * CCH + c) << 16;
      float v000 = vol[b0 + i00], v001 = vol[b0 + i01];
      float v010 = vol[b0 + i10], v011 = vol[b0 + i11];
      float v100 = vol[b1 + i00], v101 = vol[b1 + i01];
      float v110 = vol[b1 + i10], v111 = vol[b1 + i11];
      float a00 = v000 + wx * (v001 - v000);
      float a01 = v010 + wx * (v011 - v010);
      float a10 = v100 + wx * (v101 - v100);
      float a11 = v110 + wx * (v111 - v110);
      float f0 = a00 + wy * (a01 - a00);
      float f1 = a10 + wy * (a11 - a10);
      return f0 + wz * (f1 - f0);
    };

    float sj0 = tri(kv_j, c0), sj1 = tri(kv_j, c1);
    float sp0 = tri(kv_p, c0), sp1 = tri(kv_p, c1);
    sj[r][0] = sj0; sj[r][1] = sj1;
    float part = 0.125f * ((fj0 * sj0 + fj1 * sj1) + bet * (fp0 * sp0 + fp1 * sp1));
    logit[r] = wave_sum32(part);
  }

  float m  = fmaxf(logit[0], fmaxf(logit[1], logit[2]));
  float e0 = __expf(logit[0] - m), e1 = __expf(logit[1] - m), e2 = __expf(logit[2] - m);
  float inv = 1.f / (e0 + e1 + e2);
  float a0 = e0 * inv, a1 = e1 * inv, a2 = e2 * inv;

  float o0 = a0 * sj[0][0] + a1 * sj[1][0] + a2 * sj[2][0];
  float o1 = a0 * sj[0][1] + a1 * sj[1][1] + a2 * sj[2][1];

  __bf16* xc = xcat + (size_t)pix * 128;
  xc[c0]      = (__bf16)o0;
  xc[c1]      = (__bf16)o1;
  xc[64 + c0] = (__bf16)fj0;
  xc[64 + c1] = (__bf16)fj1;
}

// ---------------------------------------------------------------------------
// Kernels 3/4: 3x3 conv as implicit GEMM with v_wmma_f32_16x16x32_bf16.
// Block = 128 threads = 4 waves; wave `mt` owns out-channels [16mt,16mt+16),
// block owns 16 consecutive pixels of one row. The 3x18-pixel halo tile is
// staged into LDS once per block via GLOBAL_LOAD_ASYNC_TO_LDS_B128; halo
// slots outside the image are zero-filled with ds_store_b128 during staging,
// so the GEMM inner loop is branch-free: ds_load_b128 x2 + global_load x2 +
// v_wmma per K-step. Pixel slots padded by 8 halves (16B) -> bank stride 4.
// ---------------------------------------------------------------------------
template <int CIN, bool LEAKY>
__global__ __launch_bounds__(128) void conv3x3_wmma_kernel(
    const __bf16* __restrict__ in, const __bf16* __restrict__ wfrag,
    const float* __restrict__ bias, __bf16* __restrict__ out) {
  constexpr int KSTEPS   = CIN * 9 / 32;
  constexpr int SLOT     = CIN + 8;                 // halves per pixel slot
  constexpr int CHUNKS_PP = CIN * 2 / 16;           // 16B chunks per pixel
  constexpr int TOT_CHUNKS = 3 * 18 * CHUNKS_PP;
  __shared__ __bf16 tile[3 * 18 * SLOT] __attribute__((aligned(16)));

  int lane = threadIdx.x & 31;
  int mt   = threadIdx.x >> 5;
  int n0   = blockIdx.x << 4;
  int w0 = n0 & 255, h = (n0 >> 8) & 255, b = n0 >> 16;
  int px = lane & 15;
  int hi = lane >> 4;

  // ---- stage halo tile (rows h-1..h+1, cols w0-1..w0+16) into LDS ----
  // Valid pixels: async global->LDS copy. Invalid halo pixels: zero-fill so
  // the GEMM loop can read unconditionally (conv zero padding lives in LDS).
  unsigned lds_base = (unsigned)(uintptr_t)(&tile[0]);
  for (int chunk = threadIdx.x; chunk < TOT_CHUNKS; chunk += 128) {
    int k   = chunk & (CHUNKS_PP - 1);
    int p   = chunk / CHUNKS_PP;        // 0..53
    int col = p % 18;
    int row = p / 18;
    int hh = h + row - 1;
    int ww = w0 + col - 1;
    if (((unsigned)hh < 256u) && ((unsigned)ww < 256u)) {
      const void* g = (const void*)(in +
          (size_t)((b * 256 + hh) * 256 + ww) * CIN + k * 8);
      unsigned l = lds_base + (unsigned)((p * SLOT + k * 8) * 2);
      async_g2l_b128(l, g);
    } else {
      *reinterpret_cast<v8bf*>(&tile[p * SLOT + k * 8]) = (v8bf){};
    }
  }
  wait_asynccnt0();
  __syncthreads();

  // ---- implicit GEMM over K = 9 taps x CIN channels (branch-free) ----
  v8f acc = {};

  #pragma unroll
  for (int s = 0; s < 9; ++s) {
    int dy = s / 3, dx = s % 3;         // tap offsets (row = dy, col = px+dx)
    int p = dy * 18 + (px + dx);        // pixel slot in tile

    #pragma unroll
    for (int cc = 0; cc < CIN; cc += 32) {
      int ks = (s * CIN + cc) >> 5;
      int base = p * SLOT + cc + (hi << 3);
      v8bf lo  = *reinterpret_cast<const v8bf*>(&tile[base]);
      v8bf hi8 = *reinterpret_cast<const v8bf*>(&tile[base + 16]);
      v16bf bfrag;
      #pragma unroll
      for (int i = 0; i < 8; ++i) { bfrag[i] = lo[i]; bfrag[8 + i] = hi8[i]; }
      v16bf afrag = *reinterpret_cast<const v16bf*>(
          wfrag + ((size_t)((mt * KSTEPS + ks) * 32 + lane) << 4));
      acc = __builtin_amdgcn_wmma_f32_16x16x32_bf16(
          false, afrag, false, bfrag, (short)0, acc, false, false);
    }
  }

  // D layout: N = lane&15, VGPR i -> M = 16mt + 8*hi + i  (8 consecutive ch)
  int pixg  = n0 + px;
  int mbase = (mt << 4) + (hi << 3);
  v8bf ov;
  #pragma unroll
  for (int i = 0; i < 8; ++i) {
    float v = acc[i] + bias[mbase + i];
    if (LEAKY) v = (v >= 0.f) ? v : 0.1f * v;
    ov[i] = (__bf16)v;
  }
  *reinterpret_cast<v8bf*>(out + (size_t)pixg * 64 + mbase) = ov;
}

// ---------------------------------------------------------------------------
// Kernel 5: 1x1 conv to 3 channels, f32 NCHW output.
// ---------------------------------------------------------------------------
__global__ __launch_bounds__(256) void conv1x1_out_kernel(
    const __bf16* __restrict__ x, const float* __restrict__ Wh2,
    const float* __restrict__ bh2, float* __restrict__ out) {
  __shared__ float sw[3 * 64];
  __shared__ float sb[3];
  if (threadIdx.x < 192) sw[threadIdx.x] = Wh2[threadIdx.x];
  if (threadIdx.x < 3)   sb[threadIdx.x] = bh2[threadIdx.x];
  __syncthreads();

  int pix = blockIdx.x * 256 + threadIdx.x;
  const __bf16* xp = x + (size_t)pix * 64;
  float a0 = sb[0], a1 = sb[1], a2 = sb[2];
  #pragma unroll
  for (int j = 0; j < 64; ++j) {
    float v = (float)xp[j];
    a0 += v * sw[j];
    a1 += v * sw[64 + j];
    a2 += v * sw[128 + j];
  }
  int w = pix & 255, h = (pix >> 8) & 255, b = pix >> 16;
  int hw = (h << 8) | w;
  out[((size_t)(b * 3 + 0) << 16) + hw] = a0;
  out[((size_t)(b * 3 + 1) << 16) + hw] = a1;
  out[((size_t)(b * 3 + 2) << 16) + hw] = a2;
}

// ---------------------------------------------------------------------------
// Host launcher
// ---------------------------------------------------------------------------
extern "C" void kernel_launch(void* const* d_in, const int* in_sizes, int n_in,
                              void* d_out, int out_size, void* d_ws, size_t ws_size,
                              hipStream_t stream) {
  const float* feat_j  = (const float*)d_in[0];
  const float* feat_p  = (const float*)d_in[1];
  const float* kv_j    = (const float*)d_in[2];
  const float* kv_p    = (const float*)d_in[3];
  const float* grids   = (const float*)d_in[4];
  const float* beta    = (const float*)d_in[5];
  const float* W_aggre = (const float*)d_in[6];
  const float* b_aggre = (const float*)d_in[7];
  const float* W_h1    = (const float*)d_in[8];
  const float* b_h1    = (const float*)d_in[9];
  const float* W_h2    = (const float*)d_in[10];
  const float* b_h2    = (const float*)d_in[11];

  char* ws = (char*)d_ws;
  __bf16* xcat  = (__bf16*)(ws);                 // 33,554,432 B (NPIX*128*2)
  __bf16* featA = (__bf16*)(ws + 33554432);      // 16,777,216 B (NPIX*64*2)
  __bf16* fragA = (__bf16*)(ws + 50331648);      //    147,456 B
  __bf16* fragH = (__bf16*)(ws + 50479104);      //     73,728 B
  __bf16* xh1   = (__bf16*)(ws);                 // aliases dead xcat

  repack_weights_kernel<<<432, 256, 0, stream>>>(W_aggre, W_h1, fragA, fragH);
  attn_sample_kernel<<<NPIX / 8, 256, 0, stream>>>(feat_j, feat_p, kv_j, kv_p,
                                                   grids, beta, xcat);
  conv3x3_wmma_kernel<128, false><<<NPIX / 16, 128, 0, stream>>>(xcat, fragA,
                                                                 b_aggre, featA);
  conv3x3_wmma_kernel<64, true><<<NPIX / 16, 128, 0, stream>>>(featA, fragH,
                                                               b_h1, xh1);
  conv1x1_out_kernel<<<NPIX / 256, 256, 0, stream>>>(xh1, W_h2, b_h2,
                                                     (float*)d_out);
}
